// GCNScratch_88940182765726
// MI455X (gfx1250) — matrix operand order repaired and checked by reading, not compiled, not run on hardware
//
#include <hip/hip_runtime.h>
#include <hip/hip_bf16.h>

// GCN forward for MI455X (gfx1250, wave32).
//   N=8192 nodes, IN_F=1024, HID=256, OUT=40, E=262144 edges.
// Pipeline:
//   1) build A_tilde as a bit-matrix (8192x8192 bits = 8MB) via atomicOr (dedup == .set(1.0))
//   2) deg via popcount -> dinv = deg^-1/2
//   3) H0 = X @ W1 with split-f16 WMMA (3x v_wmma_f32_16x16x32_f16 per K-tile, ~fp32 accurate)
//   4) H1 = relu(dinv_i * sum_j dinv_j*H0[j] + b1) via bitmap scan (H0 fits in 192MB L2)
//   5) G = H1 @ W2 (40 cols padded to 48, masked) with split-f16 WMMA
//   6) out = log_softmax(dinv_i * sum_j dinv_j*G[j] + b2), wave-level reduction (wave32)

typedef __attribute__((ext_vector_type(16))) _Float16 v16h;
typedef __attribute__((ext_vector_type(8)))  float    v8f;

#define GCN_N    8192
#define GCN_INF  1024
#define GCN_HID  256
#define GCN_OUT  40
#define BMP_W    256   // 8192 bits / 32 = 256 u32 words per row

// ---------------------------------------------------------------- bitmap zero
__global__ __launch_bounds__(256) void gcn_zero_bmp(unsigned* __restrict__ bmp) {
    int idx = blockIdx.x * 256 + threadIdx.x;
    bmp[idx] = 0u;   // grid sized exactly N*BMP_W/256
}

// ------------------------------------------------- scatter edges + self loops
__global__ __launch_bounds__(256) void gcn_scatter(const int* __restrict__ ei,
                                                   unsigned* __restrict__ bmp,
                                                   int E) {
    int e = blockIdx.x * 256 + threadIdx.x;
    if (e < E) {
        int src = ei[e];        // row
        int dst = ei[E + e];    // col
        atomicOr(&bmp[src * BMP_W + (dst >> 5)], 1u << (dst & 31));
    }
    if (e < GCN_N) {            // self loop (A_tilde = A + I)
        atomicOr(&bmp[e * BMP_W + (e >> 5)], 1u << (e & 31));
    }
}

// ------------------------------------------------------------ degree^-1/2
__global__ __launch_bounds__(256) void gcn_deg(const unsigned* __restrict__ bmp,
                                               float* __restrict__ dinv) {
    int i = blockIdx.x * 256 + threadIdx.x;
    if (i < GCN_N) {
        int c = 0;
        const unsigned* row = bmp + i * BMP_W;
        #pragma unroll 8
        for (int w = 0; w < BMP_W; ++w) c += __popc(row[w]);
        dinv[i] = rsqrtf((float)c);   // c >= 1 (self loop)
    }
}

// ------------------------------------------- GEMM1: H0 = X(8192x1024) @ W1(1024x256)
// one wave per 16x16 tile; split-f16 for near-fp32 accuracy
__global__ __launch_bounds__(32) void gcn_gemm1(const float* __restrict__ X,
                                                const float* __restrict__ W1,
                                                float* __restrict__ H0) {
    const int row0 = blockIdx.x * 16;
    const int col0 = blockIdx.y * 16;
    const int lane = threadIdx.x;
    const int m    = lane & 15;
    const int hi   = lane >> 4;          // 0 for lanes 0-15, 1 for lanes 16-31

    v8f acc = {};
    for (int kk = 0; kk < GCN_INF; kk += 32) {
        // A tile (16x32 f16): halves 0..15 -> K = kk + 16*hi + h
        const float* xr = X + (row0 + m) * GCN_INF + kk + hi * 16;
        v16h ah, al;
        #pragma unroll
        for (int h = 0; h < 8; ++h) {
            float v = xr[h];
            _Float16 vh = (_Float16)v;
            ah[h] = vh; al[h] = (_Float16)(v - (float)vh);
        }
        #pragma unroll
        for (int h = 0; h < 8; ++h) {
            float v = xr[8 + h];
            _Float16 vh = (_Float16)v;
            ah[8 + h] = vh; al[8 + h] = (_Float16)(v - (float)vh);
        }
        // B tile (32x16 f16): N = m, K = kk + 16*hi + h
        const float* wr = W1 + (kk + hi * 16) * GCN_HID + col0 + m;
        v16h bh, bl;
        #pragma unroll
        for (int h = 0; h < 16; ++h) {
            float v = wr[h * GCN_HID];
            _Float16 vh = (_Float16)v;
            bh[h] = vh; bl[h] = (_Float16)(v - (float)vh);
        }
        acc = __builtin_amdgcn_wmma_f32_16x16x32_f16(false, ah, false, bh, (short)0, acc, false, false);
        acc = __builtin_amdgcn_wmma_f32_16x16x32_f16(false, ah, false, bl, (short)0, acc, false, false);
        acc = __builtin_amdgcn_wmma_f32_16x16x32_f16(false, al, false, bh, (short)0, acc, false, false);
    }
    // C/D layout: VGPR r -> row (r + 8*hi), col = m
    #pragma unroll
    for (int r = 0; r < 8; ++r)
        H0[(row0 + r + hi * 8) * GCN_HID + col0 + m] = acc[r];
}

// --------------------------- aggregation 1: H1 = relu(d_i * sum_j d_j*H0[j] + b1)
__global__ __launch_bounds__(256) void gcn_agg1(const unsigned* __restrict__ bmp,
                                                const float* __restrict__ dinv,
                                                const float* __restrict__ H0,
                                                const float* __restrict__ b1,
                                                float* __restrict__ H1) {
    __shared__ unsigned srow[BMP_W];
    const int i = blockIdx.x;
    const int f = threadIdx.x;           // feature column, 0..255
    srow[f] = bmp[i * BMP_W + f];
    __syncthreads();

    float acc = 0.0f;
    for (int w = 0; w < BMP_W; ++w) {
        unsigned word = srow[w];         // uniform across block -> uniform loop
        while (word) {
            int b = __ffs((int)word) - 1;
            word &= word - 1;
            int j = (w << 5) + b;
            acc += dinv[j] * H0[j * GCN_HID + f];
        }
    }
    float v = dinv[i] * acc + b1[f];
    H1[i * GCN_HID + f] = v > 0.0f ? v : 0.0f;
}

// ---------------------------- GEMM2: G = H1(8192x256) @ W2(256x40), cols padded to 48
__global__ __launch_bounds__(32) void gcn_gemm2(const float* __restrict__ H1,
                                                const float* __restrict__ W2,
                                                float* __restrict__ G) {
    const int row0 = blockIdx.x * 16;
    const int col0 = blockIdx.y * 16;    // blockIdx.y in 0..2 (48 padded cols)
    const int lane = threadIdx.x;
    const int m    = lane & 15;
    const int hi   = lane >> 4;
    const int col  = col0 + m;
    const bool cin = (col < GCN_OUT);

    v8f acc = {};
    for (int kk = 0; kk < GCN_HID; kk += 32) {
        const float* xr = H1 + (row0 + m) * GCN_HID + kk + hi * 16;
        v16h ah, al;
        #pragma unroll
        for (int h = 0; h < 8; ++h) {
            float v = xr[h];
            _Float16 vh = (_Float16)v;
            ah[h] = vh; al[h] = (_Float16)(v - (float)vh);
        }
        #pragma unroll
        for (int h = 0; h < 8; ++h) {
            float v = xr[8 + h];
            _Float16 vh = (_Float16)v;
            ah[8 + h] = vh; al[8 + h] = (_Float16)(v - (float)vh);
        }
        v16h bh, bl;
        #pragma unroll
        for (int h = 0; h < 16; ++h) {
            int k = kk + hi * 16 + h;
            float v = cin ? W2[k * GCN_OUT + col] : 0.0f;
            _Float16 vh = (_Float16)v;
            bh[h] = vh; bl[h] = (_Float16)(v - (float)vh);
        }
        acc = __builtin_amdgcn_wmma_f32_16x16x32_f16(false, ah, false, bh, (short)0, acc, false, false);
        acc = __builtin_amdgcn_wmma_f32_16x16x32_f16(false, ah, false, bl, (short)0, acc, false, false);
        acc = __builtin_amdgcn_wmma_f32_16x16x32_f16(false, al, false, bh, (short)0, acc, false, false);
    }
    if (cin) {
        #pragma unroll
        for (int r = 0; r < 8; ++r)
            G[(row0 + r + hi * 8) * GCN_OUT + col] = acc[r];
    }
}

// ---------- aggregation 2 + bias + log_softmax; one wave32 per node row
__global__ __launch_bounds__(32) void gcn_agg2_lsm(const unsigned* __restrict__ bmp,
                                                   const float* __restrict__ dinv,
                                                   const float* __restrict__ G,
                                                   const float* __restrict__ b2,
                                                   float* __restrict__ out) {
    const int i  = blockIdx.x;
    const int t  = threadIdx.x;          // lane owns col t, and col 32+(t&7) if t<8
    const int t2 = 32 + (t & 7);
    const bool has1 = (t < 8);

    float acc0 = 0.0f, acc1 = 0.0f;
    const unsigned* row = bmp + i * BMP_W;
    for (int w = 0; w < BMP_W; ++w) {
        unsigned word = row[w];          // same address for all lanes: broadcast
        while (word) {
            int b = __ffs((int)word) - 1;
            word &= word - 1;
            int j = (w << 5) + b;
            float dj = dinv[j];
            acc0 += dj * G[j * GCN_OUT + t];
            acc1 += dj * G[j * GCN_OUT + t2];
        }
    }
    float di = dinv[i];
    float v0 = di * acc0 + b2[t];
    float v1 = di * acc1 + b2[t2];

    // wave32 max reduction
    float m = v0;
    if (has1 && v1 > m) m = v1;
    #pragma unroll
    for (int off = 16; off > 0; off >>= 1) {
        float o = __shfl_xor(m, off, 32);
        m = o > m ? o : m;
    }
    // wave32 sum of exp
    float s = expf(v0 - m) + (has1 ? expf(v1 - m) : 0.0f);
    #pragma unroll
    for (int off = 16; off > 0; off >>= 1)
        s += __shfl_xor(s, off, 32);
    float ls = logf(s);

    out[i * GCN_OUT + t] = v0 - m - ls;
    if (has1) out[i * GCN_OUT + t2] = v1 - m - ls;
}

extern "C" void kernel_launch(void* const* d_in, const int* in_sizes, int n_in,
                              void* d_out, int out_size, void* d_ws, size_t ws_size,
                              hipStream_t stream) {
    const float* X  = (const float*)d_in[0];
    const int*   ei = (const int*)  d_in[1];   // harness delivers integers as int32
    const float* W1 = (const float*)d_in[2];
    const float* b1 = (const float*)d_in[3];
    const float* W2 = (const float*)d_in[4];
    const float* b2 = (const float*)d_in[5];
    float* out = (float*)d_out;

    const int E = in_sizes[1] / 2;

    // workspace layout (~25.3 MB total)
    char* ws = (char*)d_ws;
    unsigned* bmp  = (unsigned*)(ws);                                   // 8 MB
    float*    dinv = (float*)   (ws + (size_t)8  * 1024 * 1024);        // 32 KB
    float*    H0   = (float*)   (ws + (size_t)8  * 1024 * 1024 + 65536);// 8 MB
    float*    H1   = (float*)   (ws + (size_t)16 * 1024 * 1024 + 65536);// 8 MB
    float*    G    = (float*)   (ws + (size_t)24 * 1024 * 1024 + 65536);// 1.31 MB

    // 1) adjacency bitmap
    gcn_zero_bmp<<<GCN_N * BMP_W / 256, 256, 0, stream>>>(bmp);
    int nscatter = (E > GCN_N ? E : GCN_N);
    gcn_scatter<<<(nscatter + 255) / 256, 256, 0, stream>>>(ei, bmp, E);
    gcn_deg<<<GCN_N / 256, 256, 0, stream>>>(bmp, dinv);

    // 2) H0 = X @ W1   (WMMA, split-f16)
    gcn_gemm1<<<dim3(GCN_N / 16, GCN_HID / 16), 32, 0, stream>>>(X, W1, H0);

    // 3) H1 = relu(Â @ H0 + b1)
    gcn_agg1<<<GCN_N, 256, 0, stream>>>(bmp, dinv, H0, b1, H1);

    // 4) G = H1 @ W2   (WMMA, split-f16, 40->48 padded cols)
    gcn_gemm2<<<dim3(GCN_N / 16, 3), 32, 0, stream>>>(H1, W2, G);

    // 5) out = log_softmax(Â @ G + b2)
    gcn_agg2_lsm<<<GCN_N, 32, 0, stream>>>(bmp, dinv, G, b2, out);
}
